// TeleChatAttention_45835890983137
// MI455X (gfx1250) — compile-verified, matrix operand order checked by reading, not run on hardware
//
#include <hip/hip_runtime.h>
#include <hip/hip_bf16.h>
#include <stdint.h>

// Problem constants (match reference)
#define HSZ 4096
#define NH  32
#define HD  128
#define NB  2
#define SQ  2048
#define BS  (NB * SQ)        // 4096 token rows
#define SCALE 0.08838834764831845f            // HD^-0.5
#define CEXP  (SCALE * 1.4426950408889634f)   // scale * log2(e)

typedef __bf16 bf16;
typedef __attribute__((ext_vector_type(16))) __bf16 v16bf;
typedef __attribute__((ext_vector_type(8)))  float  v8f;

// ---------------------------------------------------------------- helpers ---

static __device__ __forceinline__ v8f wmma_bf16(v16bf a, v16bf b, v8f c) {
  // v_wmma_f32_16x16x32_bf16  (neg_a, A, neg_b, B, c_mod, C, reuse_a, reuse_b)
  return __builtin_amdgcn_wmma_f32_16x16x32_bf16(false, a, false, b, (short)0, c, false, false);
}

// Load a 16x32 bf16 A/B fragment. Memory layout: base[row][k], row pitch `pitch`
// (bf16 elements, multiple of 8 -> 16B aligned rows). Per ISA 7.12.2:
// lane L holds row = L%16, K-base kb = 8*(L/16); V0..V3 = K kb..kb+7,
// V4..V7 = K kb+16..kb+23  -> two contiguous 16-byte loads per lane.
static __device__ __forceinline__ v16bf frag_ld(const bf16* p) {
  union { v16bf f; struct { uint4 lo; uint4 hi; } u; } x;
  x.u.lo = *(const uint4*)(p);
  x.u.hi = *(const uint4*)(p + 16);
  return x.f;
}
static __device__ __forceinline__ const bf16* frag_ptr(const bf16* base, int row0,
                                                       int k0, int pitch, int lane) {
  int r  = row0 + (lane & 15);
  int kb = k0 + ((lane >> 4) << 3);
  return base + (size_t)r * pitch + kb;
}

static __device__ __forceinline__ v8f v8f_zero() {
  v8f z;
#pragma unroll
  for (int e = 0; e < 8; ++e) z[e] = 0.0f;
  return z;
}

// CDNA5 async global->LDS copy (GVS mode: 64-bit SGPR base + 32-bit VGPR byte
// offset), 16B per lane per issue, tracked by ASYNCcnt.
static __device__ __forceinline__ void async_b128(unsigned lds_byte_addr,
                                                  unsigned long long gbase,
                                                  unsigned gbyte_off) {
  asm volatile("global_load_async_to_lds_b128 %0, %1, %2"
               :
               : "v"(lds_byte_addr), "v"(gbyte_off), "s"(gbase)
               : "memory");
}
static __device__ __forceinline__ void wait_async0() {
  asm volatile("s_wait_asynccnt 0" ::: "memory");
}
static __device__ __forceinline__ unsigned lds_addr_of(const void* p) {
  return (unsigned)(uintptr_t)p;  // LDS aperture: addr[31:0] is the LDS offset
}

// --------------------------------------------------- kernel 0: f32 -> bf16 ---

__global__ __launch_bounds__(256) void cvt_bf16_kernel(const float* __restrict__ x,
                                                       bf16* __restrict__ y) {
  size_t i = ((size_t)blockIdx.x * blockDim.x + threadIdx.x) * 4;
  float4 f = *(const float4*)(x + i);
  union { bf16 b[4]; uint2 u; } p;
  p.b[0] = (bf16)f.x; p.b[1] = (bf16)f.y; p.b[2] = (bf16)f.z; p.b[3] = (bf16)f.w;
  *(uint2*)(y + i) = p.u;
}

// ------------------------------------------------ tiled WMMA GEMM (bf16) ----
// C_tile = 128x128, block = 256 threads = 8 waves (2 in M x 4 in N),
// each wave: 4x2 grid of 16x16 WMMA tiles.  K stepped by 64 (2 WMMA chunks),
// double-buffered in LDS: async-DMA A tile + VALU-converted weight tile for
// step t+1 overlap the 16 WMMAs of step t.
// MODE 0: X[BS,H] @ [Wq | Wkv] -> scatter Q,K (row-major) and V (transposed).
// MODE 1: Attn[BS,H] @ Wd + bd -> fp32 out.
#define BPITCH 72   // 64 + 8 bf16 pad -> 144B pitch, conflict-free b128 frags
#define GEMM_TILE_ELTS (128 * BPITCH)
#define GEMM_SMEM (4u * GEMM_TILE_ELTS * sizeof(bf16))  // 73,728 B

template <int MODE>
__global__ __launch_bounds__(256) void gemm_kernel(
    const bf16* __restrict__ A, const float* __restrict__ W0,
    const float* __restrict__ W1, const float* __restrict__ bias,
    float* __restrict__ outF, bf16* __restrict__ Qb, bf16* __restrict__ Kb,
    bf16* __restrict__ Vt) {
  extern __shared__ __align__(16) char smem[];
  bf16* Sbase = (bf16*)smem;  // [buf0 A][buf1 A][buf0 Bt][buf1 Bt]

  const int m0 = blockIdx.x * 128;
  const int n0 = blockIdx.y * 128;
  const int tid = threadIdx.x, lane = tid & 31, w = tid >> 5;
  const int mw = (w >> 2) * 64, nw = (w & 3) * 32;

  // weight column source for this 128-wide N tile
  const float* Wp;
  int ldw;
  if (MODE == 0) {
    if (n0 < HSZ) { Wp = W0 + n0; ldw = HSZ; }             // -> Q
    else          { Wp = W1 + (n0 - HSZ); ldw = 2 * HSZ; } // -> K or V
  } else {
    Wp = W0 + n0; ldw = HSZ;
  }

  // Per-thread staging coordinates (constant across K loop)
  const int ar  = tid >> 1;              // A-tile row
  const int ac  = (tid & 1) * 32;        // A-tile col group (bf16 elems)
  const unsigned long long aBase = (unsigned long long)(uintptr_t)A;
  const int wk  = tid >> 2;              // weight K row within tile (0..63)
  const int wng = (tid & 3) * 32;        // weight N group (0,32,64,96)

  // stage K-step tile (64 wide) into buffer `buf`
  auto stage = [&](int buf, int kt) {
    // A tile: 4 x 16B async per thread, contiguous on both sides
    unsigned aLds = lds_addr_of(Sbase + buf * GEMM_TILE_ELTS + ar * BPITCH + ac);
    unsigned goff = (unsigned)(((size_t)(m0 + ar) * HSZ + kt + ac) * sizeof(bf16));
#pragma unroll
    for (int i = 0; i < 4; ++i) async_b128(aLds + 16u * i, aBase, goff + 16u * i);
    // Bt tile: W rows kt..kt+63 (K) x cols n0..+127 (N), transpose + convert
    const float* wr = Wp + (size_t)(kt + wk) * ldw + wng;
    bf16* bs = Sbase + (2 + buf) * GEMM_TILE_ELTS;
#pragma unroll
    for (int i = 0; i < 32; i += 4) {
      float4 f = *(const float4*)(wr + i);
      bs[(wng + i + 0) * BPITCH + wk] = (bf16)f.x;
      bs[(wng + i + 1) * BPITCH + wk] = (bf16)f.y;
      bs[(wng + i + 2) * BPITCH + wk] = (bf16)f.z;
      bs[(wng + i + 3) * BPITCH + wk] = (bf16)f.w;
    }
  };

  v8f acc[4][2];
#pragma unroll
  for (int i = 0; i < 4; ++i)
#pragma unroll
    for (int j = 0; j < 2; ++j) acc[i][j] = v8f_zero();

  stage(0, 0);
  wait_async0();
  __syncthreads();

  for (int kt = 0; kt < HSZ; kt += 64) {
    const int cur = (kt >> 6) & 1;
    if (kt + 64 < HSZ) stage(cur ^ 1, kt + 64);  // overlaps with compute below

    const bf16* as = Sbase + cur * GEMM_TILE_ELTS;
    const bf16* bsc = Sbase + (2 + cur) * GEMM_TILE_ELTS;
#pragma unroll
    for (int kc = 0; kc < 2; ++kc) {
      v16bf af[4];
#pragma unroll
      for (int i = 0; i < 4; ++i)
        af[i] = frag_ld(frag_ptr(as, mw + i * 16, kc * 32, BPITCH, lane));
#pragma unroll
      for (int j = 0; j < 2; ++j) {
        v16bf bf = frag_ld(frag_ptr(bsc, nw + j * 16, kc * 32, BPITCH, lane));
#pragma unroll
        for (int i = 0; i < 4; ++i) acc[i][j] = wmma_bf16(af[i], bf, acc[i][j]);
      }
    }
    wait_async0();
    __syncthreads();
  }

  // epilogue: C layout -> element (m,n): n = lane%16 + 16j, m = v + 8*(lane/16)
#pragma unroll
  for (int i = 0; i < 4; ++i) {
#pragma unroll
    for (int j = 0; j < 2; ++j) {
      int n = n0 + nw + j * 16 + (lane & 15);
      int mbase = m0 + mw + i * 16 + ((lane >> 4) << 3);
#pragma unroll
      for (int v = 0; v < 8; ++v) {
        int m = mbase + v;
        float val = acc[i][j][v];
        if (MODE == 1) {
          outF[(size_t)m * HSZ + n] = val + bias[n];
        } else {
          int b = m >> 11, s = m & (SQ - 1);
          if (n < HSZ) {
            int h = n >> 7, d = n & 127;
            Qb[(((size_t)b * NH + h) * SQ + s) * HD + d] = (bf16)val;
          } else if (n < 2 * HSZ) {
            int n2 = n - HSZ, h = n2 >> 7, d = n2 & 127;
            Kb[(((size_t)b * NH + h) * SQ + s) * HD + d] = (bf16)val;
          } else {
            int n2 = n - 2 * HSZ, h = n2 >> 7, d = n2 & 127;
            Vt[(((size_t)b * NH + h) * HD + d) * SQ + s] = (bf16)val;  // transposed
          }
        }
      }
    }
  }
}

// --------------------------------------------------------- kernel: RoPE -----

__global__ __launch_bounds__(256) void rope_kernel(const int* __restrict__ positions,
                                                   bf16* __restrict__ Qb,
                                                   bf16* __restrict__ Kb) {
  int t = blockIdx.x * blockDim.x + threadIdx.x;  // [0, NB*NH*SQ*64)
  int j  = t & 63;
  int s  = (t >> 6) & (SQ - 1);
  int bh = t >> 17;                               // 64 * 2048 = 2^17
  int b  = bh >> 5;
  float pos  = (float)positions[b * SQ + s];
  // inv_freq = theta^(-2j/HD);  log2(10000)*2/128 = 0.2076205059304601
  float freq = __builtin_amdgcn_exp2f(-0.2076205059304601f * (float)j);
  float sn, cs;
  __sincosf(pos * freq, &sn, &cs);
  size_t base = ((size_t)bh * SQ + s) * HD + j;
  float q1 = (float)Qb[base], q2 = (float)Qb[base + 64];
  Qb[base]      = (bf16)(q1 * cs - q2 * sn);
  Qb[base + 64] = (bf16)(q2 * cs + q1 * sn);
  float k1 = (float)Kb[base], k2 = (float)Kb[base + 64];
  Kb[base]      = (bf16)(k1 * cs - k2 * sn);
  Kb[base + 64] = (bf16)(k2 * cs + k1 * sn);
}

// ---------------------------------------------- kernel: flash attention -----
// grid = (SQ/128 q-blocks, NB*NH).  256 threads = 8 waves, wave w owns q rows
// [m0, m0+16).  Online softmax over causal k-blocks of 128.  Single-buffered
// K/V staging (139 KB LDS -> 2 workgroups/WGP hide the DMA latency).
#define KPITCH 136  // 128 + 8 bf16 pad -> 272B pitch, conflict-free frags

__global__ __launch_bounds__(256) void flash_kernel(const bf16* __restrict__ Qb,
                                                    const bf16* __restrict__ Kb,
                                                    const bf16* __restrict__ Vt,
                                                    bf16* __restrict__ Attn) {
  extern __shared__ __align__(16) char smem[];
  bf16* Qs = (bf16*)smem;                 // [128][KPITCH]  Q block (persistent)
  bf16* Ks = Qs + 128 * KPITCH;           // [128][KPITCH]  K block, row = k-col
  bf16* Vs = Ks + 128 * KPITCH;           // [128][KPITCH]  V block, row = d
  bf16* Ps = Vs + 128 * KPITCH;           // [128][KPITCH]  P, per-wave strips

  const int bh = blockIdx.y;              // b*NH + h
  const int q0 = blockIdx.x * 128;
  const int tid = threadIdx.x, lane = tid & 31, w = tid >> 5;
  const int m0 = w * 16;

  const bf16* Qbh = Qb + (size_t)bh * SQ * HD;
  const bf16* Kbh = Kb + (size_t)bh * SQ * HD;
  const bf16* Vbh = Vt + (size_t)bh * HD * SQ;

  // Per-thread staging coordinates: row r, 64-elem column group c
  const int r = tid >> 1, c = (tid & 1) * 64;
  const unsigned ldsQ = lds_addr_of(Qs + r * KPITCH + c);
  const unsigned ldsK = lds_addr_of(Ks + r * KPITCH + c);
  const unsigned ldsV = lds_addr_of(Vs + r * KPITCH + c);

  // stage Q block once (async DMA: 8 x 16B per thread, contiguous both sides)
  {
    unsigned goff = (unsigned)(((size_t)(q0 + r) * HD + c) * sizeof(bf16));
#pragma unroll
    for (int i = 0; i < 8; ++i)
      async_b128(ldsQ + 16u * i, (unsigned long long)(uintptr_t)Qbh, goff + 16u * i);
  }

  v8f Oacc[8];
  float mi[8], li[8];
#pragma unroll
  for (int t = 0; t < 8; ++t) Oacc[t] = v8f_zero();
#pragma unroll
  for (int v = 0; v < 8; ++v) { mi[v] = -3.0e38f; li[v] = 0.0f; }

  const int nblocks = blockIdx.x + 1;  // causal: k-blocks 0..qblock
  for (int kb = 0; kb < nblocks; ++kb) {
    const int k0 = kb * 128;
    __syncthreads();
    { // stage K (row-major) and V (pre-transposed) via async DMA
      unsigned gk = (unsigned)(((size_t)(k0 + r) * HD + c) * sizeof(bf16));
      unsigned gv = (unsigned)(((size_t)r * SQ + k0 + c) * sizeof(bf16));
#pragma unroll
      for (int i = 0; i < 8; ++i) {
        async_b128(ldsK + 16u * i, (unsigned long long)(uintptr_t)Kbh, gk + 16u * i);
        async_b128(ldsV + 16u * i, (unsigned long long)(uintptr_t)Vbh, gv + 16u * i);
      }
    }
    wait_async0();
    __syncthreads();

    // scores: 16 q-rows x 128 k-cols, K-dim = HD
    v8f Sc[8];
#pragma unroll
    for (int nt = 0; nt < 8; ++nt) Sc[nt] = v8f_zero();
#pragma unroll
    for (int kc = 0; kc < 4; ++kc) {
      v16bf qf = frag_ld(frag_ptr(Qs, m0, kc * 32, KPITCH, lane));
#pragma unroll
      for (int nt = 0; nt < 8; ++nt) {
        v16bf bf = frag_ld(frag_ptr(Ks, nt * 16, kc * 32, KPITCH, lane));
        Sc[nt] = wmma_bf16(qf, bf, Sc[nt]);
      }
    }

    if (kb == blockIdx.x) {  // diagonal block: mask col > row (q0 == k0)
#pragma unroll
      for (int nt = 0; nt < 8; ++nt) {
        int col   = nt * 16 + (lane & 15);
        int rbase = m0 + ((lane >> 4) << 3);
#pragma unroll
        for (int v = 0; v < 8; ++v)
          if (col > rbase + v) Sc[nt][v] = -3.0e38f;
      }
    }

    // row max (per C-layout row slot v), butterfly over the 16-lane half
    float alpha[8];
#pragma unroll
    for (int v = 0; v < 8; ++v) {
      float x = Sc[0][v];
#pragma unroll
      for (int t = 1; t < 8; ++t) x = fmaxf(x, Sc[t][v]);
      x = fmaxf(x, __shfl_xor(x, 1));
      x = fmaxf(x, __shfl_xor(x, 2));
      x = fmaxf(x, __shfl_xor(x, 4));
      x = fmaxf(x, __shfl_xor(x, 8));
      float mnew = fmaxf(mi[v], x);
      alpha[v] = __builtin_amdgcn_exp2f((mi[v] - mnew) * CEXP);
      mi[v] = mnew;
    }

    // P = exp2((S - m)*c), accumulate row sums, spill to this wave's Ps strip
    float rsum[8];
#pragma unroll
    for (int v = 0; v < 8; ++v) rsum[v] = 0.0f;
#pragma unroll
    for (int nt = 0; nt < 8; ++nt) {
#pragma unroll
      for (int v = 0; v < 8; ++v) {
        float p = __builtin_amdgcn_exp2f((Sc[nt][v] - mi[v]) * CEXP);
        rsum[v] += p;
        Ps[(size_t)(m0 + v + ((lane >> 4) << 3)) * KPITCH + nt * 16 + (lane & 15)] =
            (bf16)p;
      }
    }
#pragma unroll
    for (int v = 0; v < 8; ++v) {
      float x = rsum[v];
      x += __shfl_xor(x, 1);
      x += __shfl_xor(x, 2);
      x += __shfl_xor(x, 4);
      x += __shfl_xor(x, 8);
      li[v] = li[v] * alpha[v] + x;
    }
#pragma unroll
    for (int t = 0; t < 8; ++t)
#pragma unroll
      for (int v = 0; v < 8; ++v) Oacc[t][v] *= alpha[v];

    // O += P @ V   (A = own Ps strip, B = Vs)
#pragma unroll
    for (int kc = 0; kc < 4; ++kc) {
      v16bf pf = frag_ld(frag_ptr(Ps, m0, kc * 32, KPITCH, lane));
#pragma unroll
      for (int dt = 0; dt < 8; ++dt) {
        v16bf vf = frag_ld(frag_ptr(Vs, dt * 16, kc * 32, KPITCH, lane));
        Oacc[dt] = wmma_bf16(pf, vf, Oacc[dt]);
      }
    }
  }

  // epilogue: attn[b, s, h*HD + d] as bf16 for the output GEMM
  const int h = bh & 31, b = bh >> 5;
#pragma unroll
  for (int dt = 0; dt < 8; ++dt) {
    int dcol = dt * 16 + (lane & 15);
#pragma unroll
    for (int v = 0; v < 8; ++v) {
      int row = q0 + m0 + v + ((lane >> 4) << 3);
      float o = Oacc[dt][v] * __builtin_amdgcn_rcpf(li[v]);
      size_t m = (size_t)b * SQ + row;
      Attn[m * HSZ + h * HD + dcol] = (bf16)o;
    }
  }
}

// ------------------------------------------------------------- launcher -----

extern "C" void kernel_launch(void* const* d_in, const int* in_sizes, int n_in,
                              void* d_out, int out_size, void* d_ws, size_t ws_size,
                              hipStream_t stream) {
  const int*   positions = (const int*)d_in[0];
  const float* hidden    = (const float*)d_in[1];
  const float* Wq        = (const float*)d_in[2];
  const float* Wkv       = (const float*)d_in[3];
  const float* Wd        = (const float*)d_in[4];
  const float* bd        = (const float*)d_in[5];
  float* out = (float*)d_out;

  // workspace layout (bf16): Xb | Qb | Kb | Vt | Attn, each BS*H elements
  const size_t NELT = (size_t)BS * HSZ;  // 16,777,216
  bf16* Xb   = (bf16*)d_ws;
  bf16* Qb   = Xb + NELT;
  bf16* Kb   = Qb + NELT;
  bf16* Vt   = Kb + NELT;
  bf16* Attn = Vt + NELT;

  // 1) hidden f32 -> bf16
  cvt_bf16_kernel<<<(NELT / 4 + 255) / 256, 256, 0, stream>>>(hidden, Xb);

  // 2) fused QKV projection (N = 3H), scatter to Q/K row-major, V transposed
  gemm_kernel<0><<<dim3(BS / 128, 3 * HSZ / 128), 256, GEMM_SMEM, stream>>>(
      Xb, Wq, Wkv, nullptr, nullptr, Qb, Kb, Vt);

  // 3) RoPE on Q and K (in place)
  {
    size_t pairs = (size_t)NB * NH * SQ * (HD / 2);  // 8,388,608
    rope_kernel<<<(unsigned)(pairs / 256), 256, 0, stream>>>(positions, Qb, Kb);
  }

  // 4) causal flash attention
  {
    size_t smem = 4u * 128u * KPITCH * sizeof(bf16);  // 139,264 B of 320 KB/WGP
    flash_kernel<<<dim3(SQ / 128, NB * NH), 256, smem, stream>>>(Qb, Kb, Vt, Attn);
  }

  // 5) output projection + bias -> fp32 out
  gemm_kernel<1><<<dim3(BS / 128, HSZ / 128), 256, GEMM_SMEM, stream>>>(
      Attn, Wd, nullptr, bd, out, nullptr, nullptr, nullptr);
}
